// Net_89111981457829
// MI455X (gfx1250) — matrix operand order repaired and checked by reading, not compiled, not run on hardware
//
#include <hip/hip_runtime.h>
#include <hip/hip_bf16.h>

typedef __bf16 bf16;
typedef __attribute__((ext_vector_type(16))) __bf16 v16bf;
typedef __attribute__((ext_vector_type(8)))  __bf16 v8bf;
typedef __attribute__((ext_vector_type(8)))  float  v8f;

#define TM 128
#define TN 128
#define TK 32
#define LDT 40   // padded LDS row stride in bf16 elems (80 bytes, 16B aligned, 20-bank stride)
#define BUFE ((TM + TN) * LDT)   // elems per double-buffer slot

// epilogue modes
#define EPI_LIN_F32   0
#define EPI_GATE_BF16 1
#define EPI_TANH_BF16 2
#define EPI_SIG_F32   3
#define EPI_LIN_BF16  4
#define EPI_TANH_F32  5
#define EPI_RELU_BF16 6

struct GemmDesc {
  const bf16* A; const bf16* Bt; const float* bias;
  float* outF; bf16* outB; const float* E;
  int lda, ldo, ldE, M, N, K, epi;
};
struct GemmBatch { GemmDesc d[4]; };

struct TDesc { const float* src; bf16* dst; int K; int N; int ldd; };
struct TBatch { TDesc d[24]; };

__device__ __forceinline__ float dsigmoid(float x) { return 1.0f / (1.0f + __expf(-x)); }

// async DMA: global (16B per lane) -> LDS, tracked by ASYNCcnt
__device__ __forceinline__ void async_b128(unsigned lds_off, const void* gaddr) {
  asm volatile("global_load_async_to_lds_b128 %0, %1, off"
               :: "v"(lds_off), "v"(gaddr) : "memory");
}
__device__ __forceinline__ void wait_async0() {
  asm volatile("s_wait_asynccnt 0x0" ::: "memory");
}

// ---------------------------------------------------------------------------
// Batched bf16 WMMA GEMM:  out = epi( A(MxK) * Bt(NxK)^T + bias )
// WG = 256 threads = 8 waves (4x2), tile 128x128, wave tile 32x64, K step 32.
// Double-buffered LDS; tiles DMA'd with global_load_async_to_lds_b128 so the
// next tile's fill overlaps the current tile's WMMAs.
// ---------------------------------------------------------------------------
__global__ __launch_bounds__(256) void gemm_bf16(GemmBatch batch) {
  GemmDesc g = batch.d[blockIdx.z];
  int m0 = blockIdx.x * TM;
  int n0 = blockIdx.y * TN;
  if (m0 >= g.M || n0 >= g.N) return;

  __shared__ bf16 smem[2 * BUFE];   // [buf][A 128x40 | B 128x40]

  const int tid  = threadIdx.x;
  const int lane = tid & 31;
  const int wave = tid >> 5;
  const int wm   = wave >> 1;     // 0..3  (M dir)
  const int wn   = wave & 1;      // 0..1  (N dir)
  const int l15  = lane & 15;
  const int hi   = lane >> 4;     // 0 or 1

  v8f acc[2][4] = {};

  const int lr = tid >> 1;            // 0..127 tile row for cooperative load
  const int lc = (tid & 1) * 16;      // 0 or 16 (k chunk)

  auto issue_tile = [&](int k0, int buf) {
    const int base = buf * BUFE;
    // A tile 128x32 (always in range: M is a multiple of 128 here)
    const bf16* ga = g.A + (size_t)(m0 + lr) * g.lda + k0 + lc;
    unsigned la = (unsigned)(uintptr_t)&smem[base + lr * LDT + lc];
    async_b128(la, ga);
    async_b128(la + 16u, ga + 8);
    // B tile 128x32 from Wt (N x K row-major), zero-fill rows >= N
    int nrow = n0 + lr;
    if (nrow < g.N) {
      const bf16* gb = g.Bt + (size_t)nrow * g.K + k0 + lc;
      unsigned lb = (unsigned)(uintptr_t)&smem[base + TM * LDT + lr * LDT + lc];
      async_b128(lb, gb);
      async_b128(lb + 16u, gb + 8);
    } else {
      v8bf z = {};
      *(v8bf*)&smem[base + TM * LDT + lr * LDT + lc]     = z;
      *(v8bf*)&smem[base + TM * LDT + lr * LDT + lc + 8] = z;
    }
  };

  issue_tile(0, 0);
  int buf = 0;
  for (int k0 = 0; k0 < g.K; k0 += TK) {
    wait_async0();            // current tile landed in LDS (this wave's DMAs)
    __syncthreads();          // ...and everyone else's
    if (k0 + TK < g.K) issue_tile(k0 + TK, buf ^ 1);   // overlap next fill

    const int base = buf * BUFE;
    v16bf af[2], bfr[4];
#pragma unroll
    for (int mi = 0; mi < 2; mi++) {
      int row = wm * 32 + mi * 16 + l15;   // A lane = one M row
      int kb  = hi * 8;                    // lanes<16: K 0..7 & 16..23 ; else 8..15 & 24..31
      union { v16bf v; v8bf h[2]; } u;
      u.h[0] = *(const v8bf*)&smem[base + row * LDT + kb];
      u.h[1] = *(const v8bf*)&smem[base + row * LDT + kb + 16];
      af[mi] = u.v;
    }
#pragma unroll
    for (int ni = 0; ni < 4; ni++) {
      int row = wn * 64 + ni * 16 + l15;   // B lane = one N column
      int kb  = hi * 8;                    // lanes<16: K 0..7 & 16..23 ; else 8..15 & 24..31
      union { v16bf v; v8bf h[2]; } u;
      u.h[0] = *(const v8bf*)&smem[base + TM * LDT + row * LDT + kb];
      u.h[1] = *(const v8bf*)&smem[base + TM * LDT + row * LDT + kb + 16];
      bfr[ni] = u.v;
    }
#pragma unroll
    for (int mi = 0; mi < 2; mi++)
#pragma unroll
      for (int ni = 0; ni < 4; ni++)
        acc[mi][ni] = __builtin_amdgcn_wmma_f32_16x16x32_bf16(
            false, af[mi], false, bfr[ni], (short)0, acc[mi][ni], false, false);
    __syncthreads();          // done reading buf before it is refilled
    buf ^= 1;
  }

  // ---- fused epilogue; C/D: elem e -> row e + 8*hi, col lane&15 ----
#pragma unroll
  for (int mi = 0; mi < 2; mi++) {
#pragma unroll
    for (int ni = 0; ni < 4; ni++) {
      int col = n0 + wn * 64 + ni * 16 + l15;
      if (col >= g.N) continue;
      float bia = g.bias ? g.bias[col] : 0.0f;
#pragma unroll
      for (int e = 0; e < 8; e++) {
        int row = m0 + wm * 32 + mi * 16 + e + hi * 8;
        float v = acc[mi][ni][e] + bia;
        size_t off = (size_t)row * g.ldo + col;
        switch (g.epi) {
          case EPI_LIN_F32:   g.outF[off] = v; break;
          case EPI_GATE_BF16: g.outB[off] = (bf16)(dsigmoid(v) * g.E[(size_t)row * g.ldE + col]); break;
          case EPI_TANH_BF16: g.outB[off] = (bf16)tanhf(v); break;
          case EPI_SIG_F32:   g.outF[off] = dsigmoid(v); break;
          case EPI_LIN_BF16:  g.outB[off] = (bf16)v; break;
          case EPI_TANH_F32:  g.outF[off] = tanhf(v); break;
          default:            g.outB[off] = (bf16)fmaxf(v, 0.0f); break; // relu
        }
      }
    }
  }
}

// ---------------------------------------------------------------------------
// Tiled transpose + f32 -> bf16 convert:  src (K x N) -> dst (N rows, ldd), col k
// ---------------------------------------------------------------------------
__global__ __launch_bounds__(256) void transpose_bf16(TBatch tb) {
  TDesc t = tb.d[blockIdx.z];
  __shared__ float tile[32][33];
  int kt = blockIdx.x * 32;
  int nt = blockIdx.y * 32;
  if (kt >= t.K || nt >= t.N) return;
  int tx = threadIdx.x & 31, ty = threadIdx.x >> 5;
  for (int i = ty; i < 32; i += 8) {
    int k = kt + i, n = nt + tx;
    tile[i][tx] = (k < t.K && n < t.N) ? t.src[(size_t)k * t.N + n] : 0.0f;
  }
  __syncthreads();
  for (int i = ty; i < 32; i += 8) {
    int n = nt + i, k = kt + tx;
    if (n < t.N && k < t.K) t.dst[(size_t)n * t.ldd + k] = (bf16)tile[tx][i];
  }
}

// x (B*T x 1024 f32) -> Xl (r x 512), Xa (r x 256), Xv (r x 256) bf16
__global__ void convert_x(const float* x, bf16* xl, bf16* xa, bf16* xv) {
  int idx = blockIdx.x * 256 + threadIdx.x;
  if (idx >= 256 * 48 * 1024) return;
  int r = idx >> 10, col = idx & 1023;
  float v = x[idx];
  if (col < 512)      xl[(size_t)r * 512 + col]         = (bf16)v;
  else if (col < 768) xa[(size_t)r * 256 + (col - 512)] = (bf16)v;
  else                xv[(size_t)r * 256 + (col - 768)] = (bf16)v;
}

// zero c state + activation buffers; build concatenated [zb1|gb1] bias
__global__ void init_state(float* cl, float* ca, float* cv,
                           bf16* actl, bf16* acta, bf16* actv,
                           float* biascat, const float* zb1, const float* gb1) {
  int idx = blockIdx.x * 256 + threadIdx.x;
  if (idx < 262144) { cl[idx] = 0.f; ca[idx] = 0.f; cv[idx] = 0.f; }
  if (idx < 524288) actl[idx] = (bf16)0.f;
  if (idx < 458752) { acta[idx] = (bf16)0.f; actv[idx] = (bf16)0.f; }
  if (idx < 1024)      biascat[idx] = zb1[idx];
  else if (idx < 2048) biascat[idx] = gb1[idx - 1024];
}

// LSTM pointwise: gates -> c,h ; write h as bf16 into Act at its k-slot
__global__ void lstm_elem(const float* sl, const float* sa, const float* sv,
                          float* cl, float* ca, float* cv,
                          float* hl, float* ha, float* hv,
                          bf16* actl, bf16* acta, bf16* actv) {
  int idx = blockIdx.x * 256 + threadIdx.x;      // 3 * 256 * 1024
  int mod = idx >> 18;
  int r = idx & 262143;
  int b = r >> 10, j = r & 1023;
  const float* s; float* c; float* h; bf16* act; int doff, ldact;
  if (mod == 0) { s = sl; c = cl; h = hl; act = actl; doff = 512; ldact = 2048; }
  else if (mod == 1) { s = sa; c = ca; h = ha; act = acta; doff = 256; ldact = 1792; }
  else { s = sv; c = cv; h = hv; act = actv; doff = 256; ldact = 1792; }
  const float* srow = s + (size_t)b * 4096;
  float f = srow[j], ig = srow[1024 + j], o = srow[2048 + j], ch = srow[3072 + j];
  float cn = dsigmoid(f) * c[r] + dsigmoid(ig) * tanhf(ch);
  float hn = dsigmoid(o) * tanhf(cn);
  c[r] = cn; h[r] = hn;
  act[(size_t)b * ldact + doff + j] = (bf16)hn;
}

// h_cat = [0.7*hl*attx1 + 0.3*hl*attx2 | hv*atty1 | ha*atty2]  (bf16)
__global__ void hcat_elem(const float* hl, const float* hv, const float* ha,
                          const float* attx1, const float* attx2,
                          const float* atty1, const float* atty2, bf16* hcat) {
  int idx = blockIdx.x * 256 + threadIdx.x;      // 256*3072
  int b = idx / 3072, col = idx % 3072;
  float v;
  if (col < 1024) { int j = col; float h = hl[b * 1024 + j];
    v = 0.7f * h * attx1[b * 1024 + j] + 0.3f * h * attx2[b * 1024 + j]; }
  else if (col < 2048) { int j = col - 1024; v = hv[b * 1024 + j] * atty1[b * 1024 + j]; }
  else { int j = col - 2048; v = ha[b * 1024 + j] * atty2[b * 1024 + j]; }
  hcat[idx] = (bf16)v;
}

// z = gamma * z_hat -> bf16 into all three Act z-slots; stage next x slice
__global__ void z_update(const float* zh, const float* gm,
                         bf16* actl, bf16* acta, bf16* actv,
                         const bf16* xl, const bf16* xag, const bf16* xvg,
                         int t_next, int do_z) {
  int idx = blockIdx.x * 256 + threadIdx.x;      // 131072 + 262144
  if (idx < 131072) {
    if (do_z) {
      float z = gm[idx] * zh[idx];
      int b = idx >> 9, j = idx & 511;
      bf16 zb = (bf16)z;
      actl[(size_t)b * 2048 + 1536 + j] = zb;
      acta[(size_t)b * 1792 + 1280 + j] = zb;
      actv[(size_t)b * 1792 + 1280 + j] = zb;
    }
  } else {
    int c = idx - 131072;
    if (t_next < 48) {
      int b = c >> 10, col = c & 1023;
      int r = b * 48 + t_next;
      if (col < 512)      actl[(size_t)b * 2048 + col]         = xl[(size_t)r * 512 + col];
      else if (col < 768) acta[(size_t)b * 1792 + (col - 512)] = xag[(size_t)r * 256 + (col - 512)];
      else                actv[(size_t)b * 1792 + (col - 768)] = xvg[(size_t)r * 256 + (col - 768)];
    }
  }
}

// y[b] = y1[b,:] . pW2 + pb2
__global__ void final_pred(const bf16* y1, const float* pw2, const float* pb2, float* out) {
  int b = blockIdx.x * 64 + threadIdx.x;
  if (b >= 256) return;
  float s = pb2[0];
  for (int j = 0; j < 512; j++) s += (float)y1[(size_t)b * 512 + j] * pw2[j];
  out[b] = s;
}

// ===========================================================================
extern "C" void kernel_launch(void* const* d_in, const int* in_sizes, int n_in,
                              void* d_out, int out_size, void* d_ws, size_t ws_size,
                              hipStream_t stream) {
  (void)in_sizes; (void)n_in; (void)out_size; (void)ws_size;
  const float* x    = (const float*)d_in[0];
  const float* gvW  = (const float*)d_in[1];  const float* gvb = (const float*)d_in[2];
  const float* gaW  = (const float*)d_in[3];  const float* gab = (const float*)d_in[4];
  const float* Wx_l = (const float*)d_in[5];  const float* Wh_l = (const float*)d_in[6];
  const float* Wz_l = (const float*)d_in[7];  const float* b_l  = (const float*)d_in[8];
  const float* Wx_a = (const float*)d_in[9];  const float* Wh_a = (const float*)d_in[10];
  const float* Wz_a = (const float*)d_in[11]; const float* b_a  = (const float*)d_in[12];
  const float* Wx_v = (const float*)d_in[13]; const float* Wh_v = (const float*)d_in[14];
  const float* Wz_v = (const float*)d_in[15]; const float* b_v  = (const float*)d_in[16];
  const float* c1Wx = (const float*)d_in[17]; const float* c1Wy = (const float*)d_in[18];
  const float* c1Wxk= (const float*)d_in[19]; const float* c1Wyk= (const float*)d_in[20];
  const float* c2Wx = (const float*)d_in[21]; const float* c2Wy = (const float*)d_in[22];
  const float* c2Wxk= (const float*)d_in[23]; const float* c2Wyk= (const float*)d_in[24];
  const float* zW1  = (const float*)d_in[25]; const float* zb1  = (const float*)d_in[26];
  const float* zW2  = (const float*)d_in[27]; const float* zb2  = (const float*)d_in[28];
  const float* gW1  = (const float*)d_in[29]; const float* gb1  = (const float*)d_in[30];
  const float* gW2  = (const float*)d_in[31]; const float* gb2  = (const float*)d_in[32];
  const float* pW1  = (const float*)d_in[33]; const float* pb1  = (const float*)d_in[34];
  const float* pW2  = (const float*)d_in[35]; const float* pb2  = (const float*)d_in[36];

  // ---- workspace carve-up ----
  char* p = (char*)d_ws;
  auto ab = [&](size_t elems) { bf16*  r = (bf16*)p;  p += (elems * 2 + 255) & ~(size_t)255; return r; };
  auto af = [&](size_t elems) { float* r = (float*)p; p += (elems * 4 + 255) & ~(size_t)255; return r; };

  bf16* Wl_t   = ab((size_t)4096 * 2048);
  bf16* Wa_t   = ab((size_t)4096 * 1792);
  bf16* Wv_t   = ab((size_t)4096 * 1792);
  bf16* Wzg1_t = ab((size_t)2048 * 3072);
  bf16* zW2_t  = ab((size_t)512 * 1024);
  bf16* gW2_t  = ab((size_t)512 * 1024);
  bf16* Wx1_t  = ab(64 * 1024); bf16* Wy1_t  = ab(64 * 1024);
  bf16* Wxk1_t = ab(1024 * 64); bf16* Wyk1_t = ab(1024 * 64);
  bf16* Wx2_t  = ab(64 * 1024); bf16* Wy2_t  = ab(64 * 1024);
  bf16* Wxk2_t = ab(1024 * 64); bf16* Wyk2_t = ab(1024 * 64);
  bf16* gvW_t  = ab(256 * 256); bf16* gaW_t  = ab(256 * 256);
  bf16* pW1_t  = ab(512 * 512);
  bf16* Xl_bf  = ab((size_t)12288 * 512);
  bf16* Xa_bf  = ab((size_t)12288 * 256);
  bf16* Xv_bf  = ab((size_t)12288 * 256);
  bf16* XaG    = ab((size_t)12288 * 256);
  bf16* XvG    = ab((size_t)12288 * 256);
  bf16* Act_l  = ab(256 * 2048);
  bf16* Act_a  = ab(256 * 1792);
  bf16* Act_v  = ab(256 * 1792);
  bf16* hcat   = ab(256 * 3072);
  bf16* zg1    = ab(256 * 2048);
  bf16* ax1 = ab(256 * 64); bf16* ay1 = ab(256 * 64);
  bf16* ax2 = ab(256 * 64); bf16* ay2 = ab(256 * 64);
  bf16* y1  = ab(256 * 512);
  float* s_l = af(256 * 4096); float* s_a = af(256 * 4096); float* s_v = af(256 * 4096);
  float* c_l = af(256 * 1024); float* c_a = af(256 * 1024); float* c_v = af(256 * 1024);
  float* h_l = af(256 * 1024); float* h_a = af(256 * 1024); float* h_v = af(256 * 1024);
  float* attx1 = af(256 * 1024); float* atty1 = af(256 * 1024);
  float* attx2 = af(256 * 1024); float* atty2 = af(256 * 1024);
  float* z_hat = af(256 * 512);  float* gamma = af(256 * 512);
  float* biascat = af(2048);

  // ---- 1) one-shot weight transpose/convert ----
  TBatch tb;
  int ti = 0;
  auto T_ = [&](const float* s, bf16* d, int K, int N, int ldd) { tb.d[ti++] = {s, d, K, N, ldd}; };
  T_(gvW,  gvW_t, 256, 256, 256);          T_(gaW,  gaW_t, 256, 256, 256);
  T_(Wx_l, Wl_t + 0,    512, 4096, 2048);  T_(Wh_l, Wl_t + 512,  1024, 4096, 2048);
  T_(Wz_l, Wl_t + 1536, 512, 4096, 2048);
  T_(Wx_a, Wa_t + 0,    256, 4096, 1792);  T_(Wh_a, Wa_t + 256,  1024, 4096, 1792);
  T_(Wz_a, Wa_t + 1280, 512, 4096, 1792);
  T_(Wx_v, Wv_t + 0,    256, 4096, 1792);  T_(Wh_v, Wv_t + 256,  1024, 4096, 1792);
  T_(Wz_v, Wv_t + 1280, 512, 4096, 1792);
  T_(c1Wx, Wx1_t, 1024, 64, 1024);  T_(c1Wy, Wy1_t, 1024, 64, 1024);
  T_(c1Wxk, Wxk1_t, 64, 1024, 64);  T_(c1Wyk, Wyk1_t, 64, 1024, 64);
  T_(c2Wx, Wx2_t, 1024, 64, 1024);  T_(c2Wy, Wy2_t, 1024, 64, 1024);
  T_(c2Wxk, Wxk2_t, 64, 1024, 64);  T_(c2Wyk, Wyk2_t, 64, 1024, 64);
  T_(zW1, Wzg1_t, 3072, 1024, 3072);
  T_(gW1, Wzg1_t + (size_t)1024 * 3072, 3072, 1024, 3072);
  T_(zW2, zW2_t, 1024, 512, 1024);  T_(gW2, gW2_t, 1024, 512, 1024);
  T_(pW1, pW1_t, 512, 512, 512);
  transpose_bf16<<<dim3(96, 128, 24), 256, 0, stream>>>(tb);

  // ---- 2) convert x, 3) gated-input GEMMs, 4) init state, 5) stage x(t=0) ----
  convert_x<<<dim3(49152), 256, 0, stream>>>(x, Xl_bf, Xa_bf, Xv_bf);

  GemmBatch gate{};
  gate.d[0] = {Xv_bf, gvW_t, gvb, nullptr, XvG, x + 768, 256, 256, 1024, 12288, 256, 256, EPI_GATE_BF16};
  gate.d[1] = {Xa_bf, gaW_t, gab, nullptr, XaG, x + 512, 256, 256, 1024, 12288, 256, 256, EPI_GATE_BF16};
  gemm_bf16<<<dim3(96, 2, 2), 256, 0, stream>>>(gate);

  init_state<<<dim3(2048), 256, 0, stream>>>(c_l, c_a, c_v, Act_l, Act_a, Act_v, biascat, zb1, gb1);
  z_update<<<dim3(1536), 256, 0, stream>>>(z_hat, gamma, Act_l, Act_a, Act_v,
                                           Xl_bf, XaG, XvG, 0, 0);

  // ---- per-step batches (pointers constant; t varies only in z_update) ----
  GemmBatch gA{};
  gA.d[0] = {Act_l, Wl_t, b_l, s_l, nullptr, nullptr, 2048, 4096, 0, 256, 4096, 2048, EPI_LIN_F32};
  gA.d[1] = {Act_a, Wa_t, b_a, s_a, nullptr, nullptr, 1792, 4096, 0, 256, 4096, 1792, EPI_LIN_F32};
  gA.d[2] = {Act_v, Wv_t, b_v, s_v, nullptr, nullptr, 1792, 4096, 0, 256, 4096, 1792, EPI_LIN_F32};

  GemmBatch c1{};
  c1.d[0] = {Act_l + 512, Wx1_t, nullptr, nullptr, ax1, nullptr, 2048, 64, 0, 256, 64, 1024, EPI_TANH_BF16};
  c1.d[1] = {Act_v + 256, Wy1_t, nullptr, nullptr, ay1, nullptr, 1792, 64, 0, 256, 64, 1024, EPI_TANH_BF16};
  c1.d[2] = {Act_l + 512, Wx2_t, nullptr, nullptr, ax2, nullptr, 2048, 64, 0, 256, 64, 1024, EPI_TANH_BF16};
  c1.d[3] = {Act_a + 256, Wy2_t, nullptr, nullptr, ay2, nullptr, 1792, 64, 0, 256, 64, 1024, EPI_TANH_BF16};

  GemmBatch c2{};
  c2.d[0] = {ay1, Wxk1_t, nullptr, attx1, nullptr, nullptr, 64, 1024, 0, 256, 1024, 64, EPI_SIG_F32};
  c2.d[1] = {ax1, Wyk1_t, nullptr, atty1, nullptr, nullptr, 64, 1024, 0, 256, 1024, 64, EPI_SIG_F32};
  c2.d[2] = {ay2, Wxk2_t, nullptr, attx2, nullptr, nullptr, 64, 1024, 0, 256, 1024, 64, EPI_SIG_F32};
  c2.d[3] = {ax2, Wyk2_t, nullptr, atty2, nullptr, nullptr, 64, 1024, 0, 256, 1024, 64, EPI_SIG_F32};

  GemmBatch z1b{};
  z1b.d[0] = {hcat, Wzg1_t, biascat, nullptr, zg1, nullptr, 3072, 2048, 0, 256, 2048, 3072, EPI_LIN_BF16};

  GemmBatch s2{};
  s2.d[0] = {zg1,        zW2_t, zb2, z_hat, nullptr, nullptr, 2048, 512, 0, 256, 512, 1024, EPI_TANH_F32};
  s2.d[1] = {zg1 + 1024, gW2_t, gb2, gamma, nullptr, nullptr, 2048, 512, 0, 256, 512, 1024, EPI_SIG_F32};

  // ---- sequential scan over T=48 ----
  for (int t = 0; t < 48; t++) {
    gemm_bf16<<<dim3(2, 32, 3), 256, 0, stream>>>(gA);
    lstm_elem<<<dim3(3072), 256, 0, stream>>>(s_l, s_a, s_v, c_l, c_a, c_v,
                                              h_l, h_a, h_v, Act_l, Act_a, Act_v);
    gemm_bf16<<<dim3(2, 1, 4), 256, 0, stream>>>(c1);
    gemm_bf16<<<dim3(2, 8, 4), 256, 0, stream>>>(c2);
    hcat_elem<<<dim3(3072), 256, 0, stream>>>(h_l, h_v, h_a, attx1, attx2, atty1, atty2, hcat);
    gemm_bf16<<<dim3(2, 16, 1), 256, 0, stream>>>(z1b);
    gemm_bf16<<<dim3(2, 4, 2), 256, 0, stream>>>(s2);
    z_update<<<dim3(1536), 256, 0, stream>>>(z_hat, gamma, Act_l, Act_a, Act_v,
                                             Xl_bf, XaG, XvG, t + 1, 1);
  }

  // ---- prediction head ----
  GemmBatch pr{};
  pr.d[0] = {Act_l + 1536, pW1_t, pb1, nullptr, y1, nullptr, 2048, 512, 0, 256, 512, 512, EPI_RELU_BF16};
  gemm_bf16<<<dim3(2, 4, 1), 256, 0, stream>>>(pr);
  final_pred<<<dim3(4), 64, 0, stream>>>(y1, pW2, pb2, (float*)d_out);
}
